// CustomGRU_69879117906355
// MI455X (gfx1250) — matrix operand order repaired and checked by reading, not compiled, not run on hardware
//
#include <hip/hip_runtime.h>
#include <cstdint>
#include <cstddef>

// ---------------------------------------------------------------------------
// Bidirectional 6-layer GRU for MI455X (gfx1250), wave32 + WMMA bf16.
//   T=512 B=64 I=128 H=256 L=6 D=2 G=3H=768
// Phase 1: xg = in @ w_ih^T + bias  (WMMA GEMM, fp32 accum, bf16 out)
// Phase 2: sequential scan, per-step 64x768x256 WMMA GEMM + gates,
//          w_hh slice brought into LDS via the Tensor Data Mover (TDM),
//          h ping-pong through L2, inter-WG phase barrier per timestep.
// ---------------------------------------------------------------------------

typedef __attribute__((ext_vector_type(16))) __bf16 v16bf;
typedef __attribute__((ext_vector_type(8)))  float  v8f;
typedef __attribute__((ext_vector_type(4)))  unsigned int v4u;
typedef __attribute__((ext_vector_type(8)))  int v8i;
typedef __attribute__((ext_vector_type(4)))  int v4i;

union FragU { uint4 q[2]; v16bf v; };

static constexpr int Tn = 512, Bn = 64, In = 128, Hn = 256, Ln = 6, Dn = 2;
static constexpr int Gn = 3 * Hn;              // 768
static constexpr int Mrows = Tn * Bn;          // 32768
static constexpr int NWGN = 8;                 // workgroups per direction in scan

__device__ __forceinline__ unsigned short f2bf(float f) {
  union { float f; unsigned u; } a; a.f = f;
  unsigned u = a.u;
  unsigned r = u + 0x7FFFu + ((u >> 16) & 1u);
  return (unsigned short)(r >> 16);
}
__device__ __forceinline__ float bf2f(unsigned short s) {
  union { unsigned u; float f; } a; a.u = ((unsigned)s) << 16;
  return a.f;
}

__device__ __forceinline__ v8f wmma_bf16(const FragU& a, const FragU& b, v8f c) {
  return __builtin_amdgcn_wmma_f32_16x16x32_bf16(false, a.v, false, b.v,
                                                 (short)0, c, false, false);
}

// ---------------------------------------------------------------------------
// fp32 -> bf16 raw conversion
// ---------------------------------------------------------------------------
__global__ void __launch_bounds__(256)
k_f32_to_bf16(const float* __restrict__ src, unsigned short* __restrict__ dst, int n) {
  int i = blockIdx.x * 256 + threadIdx.x;
  if (i < n) dst[i] = f2bf(src[i]);
}

// ---------------------------------------------------------------------------
// fused bias prep: biasA[ld][g] = b_ih + (g<2H ? b_hh : 0); biasN[ld][h]=b_hh_n
// ---------------------------------------------------------------------------
__global__ void __launch_bounds__(256)
k_bias(const float* __restrict__ b_ih0, const float* __restrict__ b_hh0,
       const float* __restrict__ b_ihr, const float* __restrict__ b_hhr,
       float* __restrict__ biasA, float* __restrict__ biasN) {
  int idx = blockIdx.x * 256 + threadIdx.x;
  if (idx >= Ln * Dn * Gn) return;
  int ld = idx / Gn, g = idx % Gn;
  int layer = ld >> 1, d = ld & 1;
  float bi = (layer == 0) ? b_ih0[d * Gn + g] : b_ihr[((layer - 1) * 2 + d) * Gn + g];
  float bh = (layer == 0) ? b_hh0[d * Gn + g] : b_hhr[((layer - 1) * 2 + d) * Gn + g];
  biasA[idx] = bi + (g < 2 * Hn ? bh : 0.f);
  if (g >= 2 * Hn) biasN[ld * Hn + (g - 2 * Hn)] = bh;
}

// ---------------------------------------------------------------------------
// xg GEMM:  xg[dir][m][g] = sum_k in[m][k] * W[(l,dir)][g][k] + biasA
// block tile 128x128, 8 waves, each wave 2 m-tiles x 4 n-tiles of 16x16.
// Fragments loaded directly from global: full reuse set lives in the 192MB L2.
// ---------------------------------------------------------------------------
__global__ void __launch_bounds__(256)
k_gemm_xg(const unsigned short* __restrict__ in,   // [Mrows][K] bf16
          const unsigned short* __restrict__ wih,  // slots of [G][512] bf16
          const float* __restrict__ biasA,         // slots of [G]
          unsigned short* __restrict__ xg,         // [2][Mrows][G] bf16
          int K, int layer) {
  const int tid = threadIdx.x;
  const int wave = tid >> 5, lane = tid & 31;
  const int dir = blockIdx.z;
  const int m0 = blockIdx.x * 128 + (wave >> 1) * 32;
  const int n0 = blockIdx.y * 128 + (wave & 1) * 64;
  const int hsel = lane >> 4, l15 = lane & 15;

  const unsigned short* w = wih + (size_t)(layer * 2 + dir) * (Gn * 512);
  const float* bias = biasA + (layer * 2 + dir) * Gn;
  unsigned short* xgd = xg + (size_t)dir * ((size_t)Mrows * Gn);

  v8f acc[2][4];
#pragma unroll
  for (int ni = 0; ni < 4; ++ni) {
    float b = bias[n0 + ni * 16 + l15];
    v8f a;
#pragma unroll
    for (int i = 0; i < 8; ++i) a[i] = b;
    acc[0][ni] = a; acc[1][ni] = a;
  }

  for (int kk = 0; kk < K; kk += 32) {
    FragU A0, A1;
    const unsigned short* ar0 = in + (size_t)(m0 + l15) * K + kk + hsel * 8;
    const unsigned short* ar1 = in + (size_t)(m0 + 16 + l15) * K + kk + hsel * 8;
    A0.q[0] = *(const uint4*)ar0; A0.q[1] = *(const uint4*)(ar0 + 16);
    A1.q[0] = *(const uint4*)ar1; A1.q[1] = *(const uint4*)(ar1 + 16);
#pragma unroll
    for (int ni = 0; ni < 4; ++ni) {
      FragU Bf;
      const unsigned short* br = w + (size_t)(n0 + ni * 16 + l15) * K + kk + hsel * 8;
      Bf.q[0] = *(const uint4*)br; Bf.q[1] = *(const uint4*)(br + 16);
      acc[0][ni] = wmma_bf16(A0, Bf, acc[0][ni]);
      acc[1][ni] = wmma_bf16(A1, Bf, acc[1][ni]);
    }
  }

#pragma unroll
  for (int mi = 0; mi < 2; ++mi)
#pragma unroll
    for (int ni = 0; ni < 4; ++ni)
#pragma unroll
      for (int i = 0; i < 8; ++i) {
        int m = m0 + mi * 16 + hsel * 8 + i;
        int n = n0 + ni * 16 + l15;
        xgd[(size_t)m * Gn + n] = f2bf(acc[mi][ni][i]);
      }
}

// ---------------------------------------------------------------------------
// per-direction inter-workgroup phase barrier (8 WGs), device-scope atomics
// ---------------------------------------------------------------------------
__device__ __forceinline__ void dir_barrier(unsigned* cnt, int phase) {
  __syncthreads();
  if (threadIdx.x == 0) {
    __threadfence();
    __hip_atomic_fetch_add(cnt, 1u, __ATOMIC_RELEASE, __HIP_MEMORY_SCOPE_AGENT);
    unsigned target = (unsigned)(NWGN * phase);
    while (__hip_atomic_load(cnt, __ATOMIC_ACQUIRE, __HIP_MEMORY_SCOPE_AGENT) < target)
      __builtin_amdgcn_s_sleep(1);
  }
  __syncthreads();
}

// ---------------------------------------------------------------------------
// recurrent scan: grid = 16 blocks (dir 0..1  x  8 column blocks of 32 cols)
// block = 256 threads (8 waves); wave owns 1 m-tile x 1 n-tile x 3 gates.
// w_hh slice (3 x 32 x 256 bf16) is TDM-DMA'd into LDS once (padded rows:
// 128 dwords data + 4 dwords pad == our 33-uint4 row stride) and stays
// resident for all 512 steps.
// ---------------------------------------------------------------------------
__global__ void __launch_bounds__(256)
k_scan(const unsigned short* __restrict__ xg,    // [2][Mrows][G] bf16
       const unsigned short* __restrict__ whh,   // slots of [G][H] bf16
       const float* __restrict__ biasN,          // slots of [H]
       const float* __restrict__ h0,             // [L*D][B][H] fp32
       unsigned short* __restrict__ hbf,         // [2 dirs][2 pp][B*H] bf16
       unsigned short* __restrict__ outb,        // next-layer input [Mrows][512] bf16
       float* __restrict__ outf,                 // fp32 output (last layer) or null
       float* __restrict__ hn,                   // h_n section of d_out
       unsigned* __restrict__ barriers,          // [2] counters (memset to 0)
       int layer) {
  __shared__ uint4 ldsq[3 * 32 * 33];            // [3][32][33] uint4 (8 bf16 each)

  const int tid = threadIdx.x;
  const int wave = tid >> 5, lane = tid & 31;
  const int dir = blockIdx.x >> 3;
  const int nb  = blockIdx.x & 7;
  const int c0  = nb * 32;
  const int mt  = wave >> 1;          // m-tile 0..3  (batch rows mt*16..mt*16+15)
  const int nt  = wave & 1;           // n-tile 0..1  (cols c0+nt*16 .. +15)
  const int hsel = lane >> 4, l15 = lane & 15;
  const int ld = layer * 2 + dir;

  const unsigned short* whhD = whh + (size_t)ld * (Gn * Hn);
  const unsigned short* xgd  = xg + (size_t)dir * ((size_t)Mrows * Gn);
  unsigned short* hb = hbf + dir * (2 * Bn * Hn);
  unsigned* cnt = barriers + dir;

  // ---- bring w_hh slice into LDS: rows {g*256 + c0 + r}, g=0..2, r=0..31
#if __has_builtin(__builtin_amdgcn_tensor_load_to_lds) && __has_builtin(__builtin_amdgcn_s_wait_tensorcnt)
  if (wave == 0) {
    // Tensor DMA descriptor (ISA 8.3-8.5): 3D tile X=256 elems, Y=32 rows,
    // Z=3 gates (Z stride = 256 rows); data_size=2B; LDS pad: 4 dwords after
    // every 128 dwords -> 528B row stride == ldsq's 33 uint4.
    unsigned long long ga =
        (unsigned long long)(const char*)(whhD + (size_t)c0 * Hn);
    unsigned ga_lo = __builtin_amdgcn_readfirstlane((unsigned)ga);
    unsigned ga_hi = __builtin_amdgcn_readfirstlane((unsigned)(ga >> 32));
    unsigned lds_b = __builtin_amdgcn_readfirstlane((unsigned)(size_t)(void*)&ldsq[0]);

    v4u g0;
    g0[0] = 1u;                                       // count=1, user mode
    g0[1] = lds_b;                                    // lds_addr
    g0[2] = ga_lo;                                    // global_addr[31:0]
    g0[3] = (ga_hi & 0x01FFFFFFu) | (2u << 30);       // addr[56:32] | type=2

    v8i g1;
    g1[0] = (int)((1u << 16)            // data_size = 2 bytes
                | (1u << 20)            // pad_enable
                | (6u << 22)            // pad_interval: 128 dwords
                | (3u << 25));          // pad_amount: 4 dwords
    g1[1] = (int)(256u << 16);          // tensor_dim0 = 256 (lo16 in [31:16])
    g1[2] = (int)(768u << 16);          // dim0 hi=0 | tensor_dim1=768 lo16
    g1[3] = (int)(256u << 16);          // dim1 hi=0 | tile_dim0 = 256
    g1[4] = (int)(32u | (3u << 16));    // tile_dim1 = 32, tile_dim2 = 3
    g1[5] = (int)256u;                  // tensor_dim0_stride = 256 (lo32)
    g1[6] = 0;                          // stride0 hi | stride1 lo16 (65536->0)
    g1[7] = 1;                          // tensor_dim1_stride hi: 65536>>16

    v4i g2; g2[0] = 3; g2[1] = 0; g2[2] = 0; g2[3] = 0;   // tensor_dim2 = 3
    v4i g3; g3[0] = 0; g3[1] = 0; g3[2] = 0; g3[3] = 0;

#if __has_include(<hip/amd_detail/amd_gfx1250_TDM.h>)
    v8i g4 = {};                        // 6-arg toolchain variant
    __builtin_amdgcn_tensor_load_to_lds(g0, g1, g2, g3, g4, 0);
#else
    __builtin_amdgcn_tensor_load_to_lds(g0, g1, g2, g3, 0);
#endif
    __builtin_amdgcn_s_wait_tensorcnt(0);
  }
#else
  // fallback: manual cooperative copy
#pragma unroll
  for (int i = 0; i < 12; ++i) {
    int chunk = tid + i * 256;
    int g = chunk >> 10;
    int rem = chunk & 1023;
    int r = rem >> 5;
    int qi = rem & 31;
    const uint4* src = (const uint4*)(whhD + (size_t)(g * Hn + c0 + r) * Hn) + qi;
    ldsq[(g * 32 + r) * 33 + qi] = *src;
  }
#endif

  // ---- init hidden state (fp32 in registers, bf16 to global ping buffer)
  const int c = c0 + nt * 16 + l15;        // this lane's hidden column
  const float bn = biasN[ld * Hn + c];
  float hprev[8];
#pragma unroll
  for (int i = 0; i < 8; ++i) {
    int m = mt * 16 + hsel * 8 + i;
    float h = h0[(size_t)(ld * Bn + m) * Hn + c];
    hprev[i] = h;
    hb[0 * (Bn * Hn) + m * Hn + c] = f2bf(h);
  }
  __syncthreads();                          // LDS weights + h init visible in WG
  int phase = 0;
  dir_barrier(cnt, ++phase);               // all of hbf[dir][0] visible

  for (int s = 0; s < Tn; ++s) {
    const int t = dir ? (Tn - 1 - s) : s;
    const unsigned short* hcur = hb + (s & 1) * (Bn * Hn);
    unsigned short* hnxt = hb + ((s + 1) & 1) * (Bn * Hn);

    // prefetch next step's xg rows into the cache hierarchy (global_prefetch_b8)
    if (s + 1 < Tn) {
      int t2 = dir ? (Tn - 2 - s) : (s + 1);
      size_t row2 = (size_t)t2 * Bn + mt * 16 + hsel * 8;
      __builtin_prefetch(&xgd[row2 * Gn + c], 0, 1);
      __builtin_prefetch(&xgd[row2 * Gn + Hn + c], 0, 1);
      __builtin_prefetch(&xgd[row2 * Gn + 2 * Hn + c], 0, 1);
    }

    v8f accR = {}, accZ = {}, accN = {};
    const int cl = nt * 16 + l15;          // local w_hh row within 32-col block
    const unsigned short* arow = hcur + (size_t)(mt * 16 + l15) * Hn;
#pragma unroll
    for (int kk = 0; kk < Hn; kk += 32) {
      FragU a;
      a.q[0] = *(const uint4*)(arow + kk + hsel * 16);
      a.q[1] = *(const uint4*)(arow + kk + 8 + hsel * 16);
      int kq = (kk >> 3) + hsel * 2;
      FragU br, bz, bm;
      br.q[0] = ldsq[(0 * 32 + cl) * 33 + kq]; br.q[1] = ldsq[(0 * 32 + cl) * 33 + kq + 1];
      bz.q[0] = ldsq[(1 * 32 + cl) * 33 + kq]; bz.q[1] = ldsq[(1 * 32 + cl) * 33 + kq + 1];
      bm.q[0] = ldsq[(2 * 32 + cl) * 33 + kq]; bm.q[1] = ldsq[(2 * 32 + cl) * 33 + kq + 1];
      accR = wmma_bf16(a, br, accR);
      accZ = wmma_bf16(a, bz, accZ);
      accN = wmma_bf16(a, bm, accN);
    }

#pragma unroll
    for (int i = 0; i < 8; ++i) {
      int m = mt * 16 + hsel * 8 + i;
      size_t row = (size_t)t * Bn + m;
      float xr = bf2f(xgd[row * Gn + c]);
      float xz = bf2f(xgd[row * Gn + Hn + c]);
      float xn = bf2f(xgd[row * Gn + 2 * Hn + c]);
      float r = 1.f / (1.f + __expf(-(xr + accR[i])));
      float z = 1.f / (1.f + __expf(-(xz + accZ[i])));
      float nv = tanhf(xn + r * (accN[i] + bn));
      float hh = (1.f - z) * nv + z * hprev[i];
      hprev[i] = hh;
      unsigned short h16 = f2bf(hh);
      hnxt[m * Hn + c] = h16;
      outb[row * (2 * Hn) + dir * Hn + c] = h16;
      if (outf) outf[row * (2 * Hn) + dir * Hn + c] = hh;
    }
    dir_barrier(cnt, ++phase);
  }

#pragma unroll
  for (int i = 0; i < 8; ++i) {
    int m = mt * 16 + hsel * 8 + i;
    hn[(size_t)(ld * Bn + m) * Hn + c] = hprev[i];
  }
}

// ---------------------------------------------------------------------------
// host orchestration
// ---------------------------------------------------------------------------
extern "C" void kernel_launch(void* const* d_in, const int* in_sizes, int n_in,
                              void* d_out, int out_size, void* d_ws, size_t ws_size,
                              hipStream_t stream) {
  (void)in_sizes; (void)n_in; (void)out_size; (void)ws_size;

  const float* x     = (const float*)d_in[0];
  const float* h0    = (const float*)d_in[1];
  const float* w_ih0 = (const float*)d_in[2];
  const float* w_hh0 = (const float*)d_in[3];
  const float* b_ih0 = (const float*)d_in[4];
  const float* b_hh0 = (const float*)d_in[5];
  const float* w_ihr = (const float*)d_in[6];
  const float* w_hhr = (const float*)d_in[7];
  const float* b_ihr = (const float*)d_in[8];
  const float* b_hhr = (const float*)d_in[9];

  float* outF = (float*)d_out;
  float* hnF  = outF + (size_t)Mrows * (2 * Hn);

  // ---- workspace carve-up (~191 MB)
  char* ws = (char*)d_ws;
  size_t off = 0;
  auto take = [&](size_t bytes) {
    size_t o = off;
    off = (off + bytes + 255) & ~(size_t)255;
    return o;
  };
  unsigned short* xbf   = (unsigned short*)(ws + take((size_t)Mrows * In * 2));
  unsigned short* outb0 = (unsigned short*)(ws + take((size_t)Mrows * 2 * Hn * 2));
  unsigned short* outb1 = (unsigned short*)(ws + take((size_t)Mrows * 2 * Hn * 2));
  unsigned short* wih   = (unsigned short*)(ws + take((size_t)12 * Gn * 512 * 2));
  unsigned short* whh   = (unsigned short*)(ws + take((size_t)12 * Gn * Hn * 2));
  float*          biasA = (float*)(ws + take((size_t)12 * Gn * 4));
  float*          biasN = (float*)(ws + take((size_t)12 * Hn * 4));
  unsigned short* xg    = (unsigned short*)(ws + take((size_t)2 * Mrows * Gn * 2));
  unsigned short* hbf   = (unsigned short*)(ws + take((size_t)2 * 2 * Bn * Hn * 2));
  unsigned*       bar   = (unsigned*)(ws + take(256));

  auto cvt = [&](const float* s, unsigned short* d, int n) {
    k_f32_to_bf16<<<(n + 255) / 256, 256, 0, stream>>>(s, d, n);
  };

  // ---- one-time conversions (cheap; redone per call for determinism)
  cvt(x, xbf, Mrows * In);
  for (int l = 0; l < Ln; ++l)
    for (int d = 0; d < Dn; ++d) {
      int slot = l * 2 + d;
      if (l == 0) {
        cvt(w_ih0 + (size_t)d * Gn * In, wih + (size_t)slot * Gn * 512, Gn * In);
        cvt(w_hh0 + (size_t)d * Gn * Hn, whh + (size_t)slot * Gn * Hn, Gn * Hn);
      } else {
        cvt(w_ihr + (size_t)((l - 1) * 2 + d) * Gn * 512,
            wih + (size_t)slot * Gn * 512, Gn * 512);
        cvt(w_hhr + (size_t)((l - 1) * 2 + d) * Gn * Hn,
            whh + (size_t)slot * Gn * Hn, Gn * Hn);
      }
    }
  k_bias<<<(Ln * Dn * Gn + 255) / 256, 256, 0, stream>>>(b_ih0, b_hh0, b_ihr, b_hhr,
                                                         biasA, biasN);

  // ---- layers
  for (int l = 0; l < Ln; ++l) {
    const unsigned short* in = (l == 0) ? xbf : ((l & 1) ? outb0 : outb1);
    unsigned short* outThis = (l & 1) ? outb1 : outb0;
    int K = (l == 0) ? In : 2 * Hn;

    k_gemm_xg<<<dim3(Mrows / 128, Gn / 128, 2), 256, 0, stream>>>(
        in, wih, biasA, xg, K, l);

    hipMemsetAsync(bar, 0, 2 * sizeof(unsigned), stream);

    k_scan<<<dim3(2 * NWGN), 256, 0, stream>>>(
        xg, whh, biasN, h0, hbf, outThis,
        (l == Ln - 1) ? outF : (float*)nullptr, hnF, bar, l);
  }
}
